// DMAC_VGG_Module_58617713655860
// MI455X (gfx1250) — compile-verified, matrix operand order checked by reading, not run on hardware
//
#include <hip/hip_runtime.h>
#include <hip/hip_fp16.h>

typedef __attribute__((ext_vector_type(16))) _Float16 v16h;
typedef __attribute__((ext_vector_type(16))) __bf16   v16bf;
typedef __attribute__((ext_vector_type(8)))  float    v8f;

union AFragH  { v16h  v; unsigned u[8]; };
union AFragBF { v16bf v; unsigned u[8]; };

#define TM  64
#define TN  32
#define TKK 64   // K per main-loop iteration (2 x wmma k32)

// ---- CDNA5 async global->LDS 16B copy (ASYNCcnt path), with fallback ------
__device__ __forceinline__ void async_copy16(void* lds, const void* gsrc) {
#if __has_builtin(__builtin_amdgcn_global_load_async_to_lds_b128)
    typedef __attribute__((ext_vector_type(4))) int v4i_t;
    typedef __attribute__((address_space(1))) v4i_t* glb_p;   // non-const, per builtin decl
    typedef __attribute__((address_space(3))) v4i_t* lds_p;
    __builtin_amdgcn_global_load_async_to_lds_b128(
        (glb_p)(unsigned long long)gsrc,
        (lds_p)(unsigned)(unsigned long long)lds, 0, 0);
#else
    asm volatile("global_load_async_to_lds_b128 %0, %1, off"
                 :: "v"((unsigned)(unsigned long long)lds),
                    "v"((unsigned long long)gsrc)
                 : "memory");
#endif
}

__device__ __forceinline__ void async_wait0() {
#if __has_builtin(__builtin_amdgcn_s_wait_asynccnt)
    __builtin_amdgcn_s_wait_asynccnt(0);
#else
    asm volatile("s_wait_asynccnt 0x0" ::: "memory");
#endif
}

// ---------------------------------------------------------------------------
// Implicit-GEMM conv (taps = 9 for 3x3 w/ dilation, 1 for 1x1), f16 WMMA,
// f32 accumulate, fused per-channel scale/shift (bias or folded BN) + ReLU.
// Block: 256 thr = 8 waves (4 cout-waves x 2 pixel-waves), tile 64Cout x 32pix,
// K stepped 64 per iteration = 2 wmma per wave per barrier.
// ---------------------------------------------------------------------------
__global__ __launch_bounds__(256)
void conv_wmma_kernel(const float* __restrict__ in, const __half* __restrict__ w,
                      const float* __restrict__ sc, const float* __restrict__ sh,
                      float* __restrict__ out,
                      int Cin, int H, int W, int Cout, int dil, int taps,
                      int Kpad, int relu, int acc)
{
    __shared__ alignas(16) _Float16 lA[TM * TKK];   // [cout][k]   8 KB
    __shared__ alignas(16) _Float16 lB[TN * TKK];   // [pixel][k]  4 KB

    const int HW  = H * W;
    const int K   = Cin * taps;
    const int n   = blockIdx.z;
    const int p0  = blockIdx.x * TN;
    const int m0  = blockIdx.y * TM;
    const int tid = threadIdx.x;
    const int lane = tid & 31;
    const int wave = tid >> 5;
    const int wm   = wave & 3;
    const int wn   = wave >> 2;
    const int half = lane >> 4;
    const int l16  = lane & 15;

    const float* inN = in + (size_t)n * Cin * HW;

    int kkv[8];
#pragma unroll
    for (int v = 0; v < 8; ++v)
        kkv[v] = (v < 4) ? (half * 8 + 2 * v) : (16 + half * 8 + 2 * (v - 4));

    v8f c = {};

    for (int kc = 0; kc < Kpad; kc += TKK) {
        // ---- stage A: 64 x 64 f16 weights via async global->LDS ----
        {
            int row  = tid >> 2;
            int segh = (tid & 3) * 16;       // halves
            if (m0 + row < Cout) {
                const __half* src = w + (size_t)(m0 + row) * Kpad + kc + segh;
                async_copy16(&lA[row * TKK + segh],     src);
                async_copy16(&lA[row * TKK + segh + 8], src + 8);
                if (kc + TKK < Kpad) __builtin_prefetch(src + TKK, 0, 3);
            } else {
                *(uint4*)&lA[row * TKK + segh]     = make_uint4(0, 0, 0, 0);
                *(uint4*)&lA[row * TKK + segh + 8] = make_uint4(0, 0, 0, 0);
            }
        }
        // ---- stage B: im2col 32 pixels x 64 k -> LDS [pixel][k] ----
        for (int e = tid; e < TN * TKK; e += 256) {
            int pl = e & (TN - 1);
            int kl = e >> 5;                 // 0..63
            int k  = kc + kl;
            float val = 0.f;
            if (k < K) {
                int cch = k / taps;
                int t   = k - cch * taps;
                int ry = 0, rx = 0;
                if (taps == 9) { ry = t / 3 - 1; rx = t % 3 - 1; }
                int p = p0 + pl;
                if (p < HW) {
                    int y = p / W + ry * dil;
                    int x = p % W + rx * dil;
                    if (y >= 0 && y < H && x >= 0 && x < W)
                        val = inN[(size_t)cch * HW + (size_t)y * W + x];
                }
            }
            lB[pl * TKK + kl] = (_Float16)val;
        }
        async_wait0();
        __syncthreads();

        const unsigned* lAu = (const unsigned*)lA;
        const unsigned* lBu = (const unsigned*)lB;
        const int arow = wm * 16 + l16;
        const int brow = wn * 16 + l16;
#pragma unroll
        for (int kh = 0; kh < 2; ++kh) {
            const int ko = kh * 16;          // u32 offset (32 halves)
            AFragH a, b;
#pragma unroll
            for (int v = 0; v < 8; ++v) {
                a.u[v] = lAu[arow * (TKK / 2) + ko + (kkv[v] >> 1)];
                b.u[v] = lBu[brow * (TKK / 2) + ko + (kkv[v] >> 1)];
            }
            c = __builtin_amdgcn_wmma_f32_16x16x32_f16(false, a.v, false, b.v,
                                                       (short)0, c, false, false);
        }
        __syncthreads();
    }

#pragma unroll
    for (int r = 0; r < 8; ++r) {
        int m = m0 + wm * 16 + r + 8 * half;
        int p = p0 + wn * 16 + l16;
        if (m < Cout && p < HW) {
            float v = c[r] * sc[m] + sh[m];
            if (relu) v = fmaxf(v, 0.f);
            size_t o = (size_t)n * Cout * HW + (size_t)m * HW + p;
            if (acc) out[o] += v; else out[o] = v;
        }
    }
}

// ---------------------------------------------------------------------------
// Correlation GEMM: M[b,i,j] = sum_c AT[b,i,c] * BT[b,j,c], bf16 WMMA.
// AT/BT are bf16 [bs][P][C]. Tile 64 i x 32 j, K stepped 64 (2 wmma/iter).
// ---------------------------------------------------------------------------
__global__ __launch_bounds__(256)
void corr_wmma_kernel(const unsigned short* __restrict__ AT,
                      const unsigned short* __restrict__ BT,
                      float* __restrict__ M, int C, int P)
{
    __shared__ alignas(16) unsigned short lA[TM * TKK];  // 8 KB
    __shared__ alignas(16) unsigned short lB[TN * TKK];  // 4 KB

    const int bb  = blockIdx.z;
    const int i0  = blockIdx.y * TM;
    const int j0  = blockIdx.x * TN;
    const int tid = threadIdx.x;
    const int lane = tid & 31;
    const int wave = tid >> 5;
    const int wm = wave & 3;
    const int wn = wave >> 2;
    const int half = lane >> 4;
    const int l16  = lane & 15;

    const unsigned short* Ab = AT + (size_t)bb * P * C;
    const unsigned short* Bb = BT + (size_t)bb * P * C;

    int kkv[8];
#pragma unroll
    for (int v = 0; v < 8; ++v)
        kkv[v] = (v < 4) ? (half * 8 + 2 * v) : (16 + half * 8 + 2 * (v - 4));

    v8f c = {};

    for (int kc = 0; kc < C; kc += TKK) {
        {   // A: 64 rows x 64 bf16, 2 async b128 per thread
            int row  = tid >> 2;
            int segh = (tid & 3) * 16;
            const unsigned short* src = &Ab[(size_t)(i0 + row) * C + kc + segh];
            async_copy16(&lA[row * TKK + segh],     src);
            async_copy16(&lA[row * TKK + segh + 8], src + 8);
        }
        {   // B: 32 rows x 64 bf16, 1 async b128 per thread
            int row  = tid >> 3;
            int segh = (tid & 7) * 8;
            async_copy16(&lB[row * TKK + segh],
                         &Bb[(size_t)(j0 + row) * C + kc + segh]);
        }
        async_wait0();
        __syncthreads();

        const unsigned* lAu = (const unsigned*)lA;
        const unsigned* lBu = (const unsigned*)lB;
        const int arow = wm * 16 + l16;
        const int brow = wn * 16 + l16;
#pragma unroll
        for (int kh = 0; kh < 2; ++kh) {
            const int ko = kh * 16;
            AFragBF a, b;
#pragma unroll
            for (int v = 0; v < 8; ++v) {
                a.u[v] = lAu[arow * (TKK / 2) + ko + (kkv[v] >> 1)];
                b.u[v] = lBu[brow * (TKK / 2) + ko + (kkv[v] >> 1)];
            }
            c = __builtin_amdgcn_wmma_f32_16x16x32_bf16(false, a.v, false, b.v,
                                                        (short)0, c, false, false);
        }
        __syncthreads();
    }

#pragma unroll
    for (int r = 0; r < 8; ++r) {
        int i = i0 + wm * 16 + r + 8 * half;
        int j = j0 + wn * 16 + l16;
        M[((size_t)bb * P + i) * P + j] = c[r];
    }
}

// ------------------------------- helpers -----------------------------------
__global__ void copy_k(const float* __restrict__ a, float* __restrict__ b, long n) {
    long i = (long)blockIdx.x * 256 + threadIdx.x;
    if (i < n) b[i] = a[i];
}

__global__ void pack_w16(const float* __restrict__ w32, __half* __restrict__ w16,
                         int Cout, int K, int Kpad) {
    long i = (long)blockIdx.x * 256 + threadIdx.x;
    long total = (long)Cout * Kpad;
    if (i >= total) return;
    int o = (int)(i / Kpad), k = (int)(i - (long)o * Kpad);
    w16[i] = (k < K) ? __float2half(w32[(size_t)o * K + k]) : __float2half(0.f);
}

__global__ void bias_ss(const float* __restrict__ bias, float* sc, float* sh, int C) {
    int i = blockIdx.x * 256 + threadIdx.x;
    if (i < C) { sc[i] = 1.f; sh[i] = bias[i]; }
}

__global__ void bn_ss(const float* __restrict__ bias, const float* __restrict__ g,
                      const float* __restrict__ bt, const float* __restrict__ mn,
                      const float* __restrict__ vr, float* sc, float* sh, int C) {
    int i = blockIdx.x * 256 + threadIdx.x;
    if (i < C) {
        float s = g[i] * rsqrtf(vr[i] + 1e-5f);
        sc[i] = s;
        sh[i] = (bias[i] - mn[i]) * s + bt[i];
    }
}

__global__ void maxpool2_k(const float* __restrict__ in, float* __restrict__ out,
                           int NC, int H, int W) {
    int Ho = H >> 1, Wo = W >> 1;
    long total = (long)NC * Ho * Wo;
    long i = (long)blockIdx.x * 256 + threadIdx.x;
    if (i >= total) return;
    int wo = (int)(i % Wo); long t = i / Wo;
    int ho = (int)(t % Ho); int nc = (int)(t / Ho);
    const float* p = in + ((long)nc * H + ho * 2) * W + wo * 2;
    out[i] = fmaxf(fmaxf(p[0], p[1]), fmaxf(p[W], p[W + 1]));
}

__global__ void maxpool3_k(const float* __restrict__ in, float* __restrict__ out,
                           int NC, int H, int W) {
    long total = (long)NC * H * W;
    long i = (long)blockIdx.x * 256 + threadIdx.x;
    if (i >= total) return;
    int x = (int)(i % W); long t = i / W;
    int y = (int)(t % H); int nc = (int)(t / H);
    const float* p = in + (long)nc * H * W;
    float v = -3.4e38f;
    for (int dy = -1; dy <= 1; ++dy)
        for (int dx = -1; dx <= 1; ++dx) {
            int yy = y + dy, xx = x + dx;
            if (yy >= 0 && yy < H && xx >= 0 && xx < W)
                v = fmaxf(v, p[(long)yy * W + xx]);
        }
    out[i] = v;
}

__global__ void pack_bf16T_k(const float* __restrict__ x, unsigned short* __restrict__ xt,
                             int C, int P, int bs) {
    long total = (long)bs * P * C;
    long i = (long)blockIdx.x * 256 + threadIdx.x;
    if (i >= total) return;
    int c = (int)(i % C); long t = i / C;
    int p = (int)(t % P); int b = (int)(t / P);
    unsigned u = __float_as_uint(x[((long)b * C + c) * P + p]);
    xt[i] = (unsigned short)((u + 0x7FFFu + ((u >> 16) & 1u)) >> 16);
}

__global__ void remap_k(const float* __restrict__ M, const int* __restrict__ idx,
                        float* __restrict__ out, int P, float invC) {
    long PP = (long)P * P;
    long i = (long)blockIdx.x * 256 + threadIdx.x;
    if (i >= 2 * PP) return;
    long q = i % PP; int b = (int)(i / PP);
    out[i] = M[b * PP + idx[q]] * invC;
}

__global__ void chansum_k(const float* __restrict__ rm, float* __restrict__ sums, int P) {
    int i = blockIdx.x * 256 + threadIdx.x;
    if (i >= 2 * P) return;
    const float* p = rm + (long)i * P;
    float s = 0.f;
    for (int t = 0; t < P; ++t) s += p[t];
    sums[i] = s;
}

__global__ void top6_k(const float* __restrict__ sums, int* __restrict__ idx, int P) {
    __shared__ float s[1024];
    int b = blockIdx.x;
    for (int t = threadIdx.x; t < P; t += blockDim.x) s[t] = sums[b * P + t];
    __syncthreads();
    if (threadIdx.x == 0) {
        int taken[6];
        for (int k = 0; k < 6; ++k) {
            float best = -3.4e38f; int bi = 0;
            for (int t = 0; t < P; ++t) {
                bool used = false;
                for (int u = 0; u < k; ++u) if (taken[u] == t) used = true;
                if (!used && s[t] > best) { best = s[t]; bi = t; }
            }
            taken[k] = bi;
            idx[b * 6 + k] = bi;
        }
    }
}

__global__ void maxavg_k(const float* __restrict__ rm, float* __restrict__ dst,
                         int P, int chBase, int Cdst) {
    int i = blockIdx.x * 256 + threadIdx.x;
    if (i >= 2 * P) return;
    int b = i / P, pix = i % P;
    const float* p = rm + (long)b * P * P + pix;
    float mx = -3.4e38f, sm = 0.f;
    for (int ch = 0; ch < P; ++ch) {
        float v = p[(long)ch * P];
        mx = fmaxf(mx, v);
        sm += v;
    }
    dst[((long)b * Cdst + chBase) * P + pix]     = mx;
    dst[((long)b * Cdst + chBase + 1) * P + pix] = sm / (float)P;
}

__global__ void sel_k(const float* __restrict__ rm, const int* __restrict__ idx,
                      float* __restrict__ dst, int P, int chBase, int Cdst) {
    long total = 2L * 6 * P;
    long i = (long)blockIdx.x * 256 + threadIdx.x;
    if (i >= total) return;
    int pix = (int)(i % P); long t = i / P;
    int k = (int)(t % 6); int b = (int)(t / 6);
    dst[((long)b * Cdst + chBase + 2 + k) * P + pix] =
        rm[((long)b * P + idx[b * 6 + k]) * P + pix];
}

// ------------------------------ orchestration ------------------------------
static void run_conv(hipStream_t s, const float* in, const float* W, const float* bias,
                     const float* g, const float* be, const float* mn, const float* vr,
                     float* out, __half* w16, float* sc, float* sh,
                     int Cin, int H, int Wd, int Cout, int dil, int taps,
                     int N, int relu, int acc)
{
    int K = Cin * taps, Kpad = (K + 63) & ~63;
    long tw = (long)Cout * Kpad;
    pack_w16<<<dim3((unsigned)((tw + 255) / 256)), dim3(256), 0, s>>>(W, w16, Cout, K, Kpad);
    if (g) bn_ss<<<(Cout + 255) / 256, 256, 0, s>>>(bias, g, be, mn, vr, sc, sh, Cout);
    else   bias_ss<<<(Cout + 255) / 256, 256, 0, s>>>(bias, sc, sh, Cout);
    int HW = H * Wd;
    conv_wmma_kernel<<<dim3((HW + TN - 1) / TN, (Cout + TM - 1) / TM, N), 256, 0, s>>>(
        in, w16, sc, sh, out, Cin, H, Wd, Cout, dil, taps, Kpad, relu, acc);
}

static void do_corr(hipStream_t s, const unsigned short* AT, const unsigned short* BT,
                    int C, float* Mbuf, float* RM, const int* nidx,
                    float* sums, int* tidx, float* dst, int chBase)
{
    const int P = 1024;
    corr_wmma_kernel<<<dim3(P / TN, P / TM, 2), 256, 0, s>>>(AT, BT, Mbuf, C, P);
    long tot = 2L * P * P;
    remap_k<<<dim3((unsigned)((tot + 255) / 256)), 256, 0, s>>>(Mbuf, nidx, RM, P, 1.f / C);
    chansum_k<<<(2 * P + 255) / 256, 256, 0, s>>>(RM, sums, P);
    top6_k<<<2, 256, 0, s>>>(sums, tidx, P);
    maxavg_k<<<(2 * P + 255) / 256, 256, 0, s>>>(RM, dst, P, chBase, 48);
    sel_k<<<(2 * 6 * P + 255) / 256, 256, 0, s>>>(RM, tidx, dst, P, chBase, 48);
}

extern "C" void kernel_launch(void* const* d_in, const int* in_sizes, int n_in,
                              void* d_out, int out_size, void* d_ws, size_t ws_size,
                              hipStream_t stream)
{
    (void)in_sizes; (void)n_in; (void)out_size; (void)ws_size;
    const float* x1 = (const float*)d_in[0];
    const float* x2 = (const float*)d_in[1];
    const int*   nidx = (const int*)d_in[84];
    auto F = [&](int i) { return (const float*)d_in[i]; };

    char* ws = (char*)d_ws;
    size_t off = 0;
    auto carve = [&](size_t bytes) {
        void* p = ws + off;
        off = (off + bytes + 255) & ~(size_t)255;
        return p;
    };
    float* A0  = (float*)carve(4ull * 64 * 65536 * 4);
    float* A1  = (float*)carve(4ull * 64 * 65536 * 4);
    float* X3  = (float*)carve(4ull * 256 * 1024 * 4);
    float* X4  = (float*)carve(4ull * 512 * 1024 * 4);
    float* X5  = (float*)carve(4ull * 512 * 1024 * 4);
    __half* W16 = (__half*)carve(512ull * 4608 * 2);
    float* SC  = (float*)carve(512 * 4);
    float* SH  = (float*)carve(512 * 4);
    unsigned short* X1T = (unsigned short*)carve(2ull * 1024 * 512 * 2);
    unsigned short* X2T = (unsigned short*)carve(2ull * 1024 * 512 * 2);
    float* MC  = (float*)carve(2ull * 1024 * 1024 * 4);
    float* RM  = (float*)carve(2ull * 1024 * 1024 * 4);
    float* SUMS = (float*)carve(2 * 1024 * 4);
    int*   TIDX = (int*)carve(2 * 6 * 4);
    float* C1  = (float*)carve(2ull * 48 * 1024 * 4);
    float* C2  = (float*)carve(2ull * 48 * 1024 * 4);
    float* T0  = (float*)carve(2ull * 96 * 1024 * 4);
    float* T1  = (float*)carve(2ull * 96 * 1024 * 4);

    long nIn = 2L * 3 * 65536;
    copy_k<<<dim3((unsigned)((nIn + 255) / 256)), 256, 0, stream>>>(x1, A0, nIn);
    copy_k<<<dim3((unsigned)((nIn + 255) / 256)), 256, 0, stream>>>(x2, A0 + nIn, nIn);

    // ---- scale3 ----
    run_conv(stream, A0, F(2),  F(3),  0,0,0,0, A1, W16, SC, SH,   3, 256, 256,  64, 1, 9, 4, 1, 0);
    run_conv(stream, A1, F(4),  F(5),  0,0,0,0, A0, W16, SC, SH,  64, 256, 256,  64, 1, 9, 4, 1, 0);
    maxpool2_k<<<dim3((unsigned)((4L*64*128*128 + 255)/256)), 256, 0, stream>>>(A0, A1, 4*64, 256, 256);
    run_conv(stream, A1, F(6),  F(7),  0,0,0,0, A0, W16, SC, SH,  64, 128, 128, 128, 1, 9, 4, 1, 0);
    run_conv(stream, A0, F(8),  F(9),  0,0,0,0, A1, W16, SC, SH, 128, 128, 128, 128, 1, 9, 4, 1, 0);
    maxpool2_k<<<dim3((unsigned)((4L*128*64*64 + 255)/256)), 256, 0, stream>>>(A1, A0, 4*128, 128, 128);
    run_conv(stream, A0, F(10), F(11), 0,0,0,0, A1, W16, SC, SH, 128,  64,  64, 256, 1, 9, 4, 1, 0);
    run_conv(stream, A1, F(12), F(13), 0,0,0,0, A0, W16, SC, SH, 256,  64,  64, 256, 1, 9, 4, 1, 0);
    run_conv(stream, A0, F(14), F(15), 0,0,0,0, A1, W16, SC, SH, 256,  64,  64, 256, 1, 9, 4, 1, 0);
    maxpool2_k<<<dim3((unsigned)((4L*256*32*32 + 255)/256)), 256, 0, stream>>>(A1, X3, 4*256, 64, 64);

    // ---- corr @ scale3 (C=256) ----
    {
        int C = 256;
        long tp = 2L * 1024 * C;
        pack_bf16T_k<<<dim3((unsigned)((tp + 255) / 256)), 256, 0, stream>>>(X3,                 X1T, C, 1024, 2);
        pack_bf16T_k<<<dim3((unsigned)((tp + 255) / 256)), 256, 0, stream>>>(X3 + 2L * C * 1024, X2T, C, 1024, 2);
        do_corr(stream, X2T, X1T, C, MC, RM, nidx, SUMS, TIDX, C1, 0);
        do_corr(stream, X1T, X1T, C, MC, RM, nidx, SUMS, TIDX, C1, 8);
        do_corr(stream, X1T, X2T, C, MC, RM, nidx, SUMS, TIDX, C2, 0);
        do_corr(stream, X2T, X2T, C, MC, RM, nidx, SUMS, TIDX, C2, 8);
    }

    // ---- scale4 ----
    run_conv(stream, X3, F(16), F(17), 0,0,0,0, A0, W16, SC, SH, 256, 32, 32, 512, 1, 9, 4, 1, 0);
    run_conv(stream, A0, F(18), F(19), 0,0,0,0, A1, W16, SC, SH, 512, 32, 32, 512, 1, 9, 4, 1, 0);
    run_conv(stream, A1, F(20), F(21), 0,0,0,0, A0, W16, SC, SH, 512, 32, 32, 512, 1, 9, 4, 1, 0);
    maxpool3_k<<<dim3((unsigned)((4L*512*32*32 + 255)/256)), 256, 0, stream>>>(A0, X4, 4*512, 32, 32);

    // ---- corr @ scale4 (C=512) ----
    {
        int C = 512;
        long tp = 2L * 1024 * C;
        pack_bf16T_k<<<dim3((unsigned)((tp + 255) / 256)), 256, 0, stream>>>(X4,                 X1T, C, 1024, 2);
        pack_bf16T_k<<<dim3((unsigned)((tp + 255) / 256)), 256, 0, stream>>>(X4 + 2L * C * 1024, X2T, C, 1024, 2);
        do_corr(stream, X2T, X1T, C, MC, RM, nidx, SUMS, TIDX, C1, 16);
        do_corr(stream, X1T, X1T, C, MC, RM, nidx, SUMS, TIDX, C1, 24);
        do_corr(stream, X1T, X2T, C, MC, RM, nidx, SUMS, TIDX, C2, 16);
        do_corr(stream, X2T, X2T, C, MC, RM, nidx, SUMS, TIDX, C2, 24);
    }

    // ---- scale5 (dilation 2) ----
    run_conv(stream, X4, F(22), F(23), 0,0,0,0, A0, W16, SC, SH, 512, 32, 32, 512, 2, 9, 4, 1, 0);
    run_conv(stream, A0, F(24), F(25), 0,0,0,0, A1, W16, SC, SH, 512, 32, 32, 512, 2, 9, 4, 1, 0);
    run_conv(stream, A1, F(26), F(27), 0,0,0,0, A0, W16, SC, SH, 512, 32, 32, 512, 2, 9, 4, 1, 0);
    maxpool3_k<<<dim3((unsigned)((4L*512*32*32 + 255)/256)), 256, 0, stream>>>(A0, X5, 4*512, 32, 32);

    // ---- corr @ scale5 (C=512) ----
    {
        int C = 512;
        long tp = 2L * 1024 * C;
        pack_bf16T_k<<<dim3((unsigned)((tp + 255) / 256)), 256, 0, stream>>>(X5,                 X1T, C, 1024, 2);
        pack_bf16T_k<<<dim3((unsigned)((tp + 255) / 256)), 256, 0, stream>>>(X5 + 2L * C * 1024, X2T, C, 1024, 2);
        do_corr(stream, X2T, X1T, C, MC, RM, nidx, SUMS, TIDX, C1, 32);
        do_corr(stream, X1T, X1T, C, MC, RM, nidx, SUMS, TIDX, C1, 40);
        do_corr(stream, X1T, X2T, C, MC, RM, nidx, SUMS, TIDX, C2, 32);
        do_corr(stream, X2T, X2T, C, MC, RM, nidx, SUMS, TIDX, C2, 40);
    }

    // ---- classifier heads ----
    const int DIL[4] = {6, 12, 18, 24};
    float* outs[2] = { (float*)d_out, (float*)d_out + 4096 };
    const float* srcs[2] = { C1, C2 };
    for (int oi = 0; oi < 2; ++oi) {
        for (int br = 0; br < 4; ++br) {
            int base = 28 + 14 * br;
            int d = DIL[br];
            run_conv(stream, srcs[oi], F(base), F(base + 1),
                     F(base + 2), F(base + 3), F(base + 4), F(base + 5),
                     T0, W16, SC, SH, 48, 32, 32, 96, d, 9, 2, 1, 0);
            run_conv(stream, T0, F(base + 6), F(base + 7),
                     F(base + 8), F(base + 9), F(base + 10), F(base + 11),
                     T1, W16, SC, SH, 96, 32, 32, 96, 1, 1, 2, 1, 0);
            run_conv(stream, T1, F(base + 12), F(base + 13),
                     0, 0, 0, 0,
                     outs[oi], W16, SC, SH, 96, 32, 32, 2, 1, 1, 2, 0, br > 0);
        }
    }
}
